// MultiHeadKANAttention_4801773437389
// MI455X (gfx1250) — matrix operand order repaired and checked by reading, not compiled, not run on hardware
//
#include <hip/hip_runtime.h>
#include <hip/hip_bf16.h>

// ---------------------------------------------------------------------------
// CDNA5 async-to-LDS path detection (gfx1250). Host pass and older toolchains
// fall back to a synchronous LDS staging copy.
// ---------------------------------------------------------------------------
#if defined(__has_builtin)
# if __has_builtin(__builtin_amdgcn_global_load_async_to_lds_b128)
#  define ASYNC_LDS 1
# endif
#endif

typedef __bf16  v16bf __attribute__((ext_vector_type(16)));
typedef float   v8f   __attribute__((ext_vector_type(8)));
typedef int     v4i_vs __attribute__((vector_size(16)));   // matches builtin param

union FragBF { v16bf v; unsigned int u[8]; unsigned short s[16]; };
union AccF   { v8f   v; float f[8]; };

__device__ __forceinline__ v8f wmma_bf16(v16bf a, v16bf b, v8f c) {
    return __builtin_amdgcn_wmma_f32_16x16x32_bf16(
        false, a, false, b, (short)0, c, false, false);
}

// Stage 16 bytes global -> LDS (per thread). Async on CDNA5 (ASYNCcnt).
__device__ __forceinline__ void stage16(__bf16* l, const __bf16* g) {
#if defined(ASYNC_LDS)
    __builtin_amdgcn_global_load_async_to_lds_b128(
        (__attribute__((address_space(1))) v4i_vs*)g,
        (__attribute__((address_space(3))) v4i_vs*)l,
        0, 0);
#else
    *(int4*)l = *(const int4*)g;
#endif
}

__device__ __forceinline__ void async_wait0() {
#if defined(ASYNC_LDS)
# if __has_builtin(__builtin_amdgcn_s_wait_asynccnt)
    __builtin_amdgcn_s_wait_asynccnt(0);
# else
    asm volatile("s_wait_asynccnt 0" ::: "memory");
# endif
#endif
}

// ---------------------------------------------------------------------------
// Fragment loaders (wave32), contiguous k order: element s of lane half h is k = 16h + s
// ---------------------------------------------------------------------------
__device__ __forceinline__ v16bf load_a_frag(const __bf16* A, int lda, int m0, int k0) {
    int lane = threadIdx.x & 31;
    int m = lane & 15, h = lane >> 4;
    const __bf16* row = A + (size_t)(m0 + m) * lda + k0;
    FragBF f;
#pragma unroll
    for (int vi = 0; vi < 8; ++vi) {
        int kk  = h * 16 + vi * 2;
        f.u[vi] = *(const unsigned int*)(row + kk);
    }
    return f.v;
}

// B[k][n] = W[n][k], W row-major (N x K): contiguous 16-element k-runs.
__device__ __forceinline__ v16bf load_b_fragT(const __bf16* W, int ldw, int n0, int k0) {
    int lane = threadIdx.x & 31;
    int n = lane & 15, h = lane >> 4;
    const __bf16* row = W + (size_t)(n0 + n) * ldw + k0 + h * 16;
    FragBF f;
#pragma unroll
    for (int vi = 0; vi < 8; ++vi)
        f.u[vi] = *(const unsigned int*)(row + vi * 2);
    return f.v;
}

// B[k][n] = V[k][n], V row-major (K x N): strided k (for P*V).
__device__ __forceinline__ v16bf load_b_fragN(const __bf16* V, int ldv, int k0, int n0) {
    int lane = threadIdx.x & 31;
    int n = lane & 15, h = lane >> 4;
    FragBF f;
#pragma unroll
    for (int vi = 0; vi < 8; ++vi) {
        int k = k0 + h * 16 + vi * 2;
        f.s[2 * vi]     = *(const unsigned short*)(V + (size_t)k       * ldv + n0 + n);
        f.s[2 * vi + 1] = *(const unsigned short*)(V + (size_t)(k + 1) * ldv + n0 + n);
    }
    return f.v;
}

// ---------------------------------------------------------------------------
#define SEQ     2048
#define HID     1024
#define OUTF    3072
#define NCOEF   8
#define KCAT    9216
#define NHEAD   16
#define HDIM    64

// ---------------------------------------------------------------------------
// 1) Pack [base_weight | spline_weight*scaler] -> bf16 Wcat (3072 x 9216)
// ---------------------------------------------------------------------------
__global__ void pack_wcat_kernel(const float* __restrict__ bw,
                                 const float* __restrict__ sw,
                                 const float* __restrict__ scaler,
                                 __bf16* __restrict__ wcat) {
    size_t i = (size_t)blockIdx.x * 256 + threadIdx.x;
    if (i >= (size_t)OUTF * KCAT) return;
    int nrow = (int)(i / KCAT);
    int col  = (int)(i % KCAT);
    float v;
    if (col < HID) {
        v = bw[(size_t)nrow * HID + col];
    } else {
        int idx = col - HID;
        int f   = idx >> 3;
        v = sw[(size_t)nrow * (HID * NCOEF) + idx] * scaler[(size_t)nrow * HID + f];
    }
    wcat[i] = (__bf16)v;
}

__global__ void pack_bf16_kernel(const float* __restrict__ src, __bf16* __restrict__ dst, int n) {
    int i = blockIdx.x * 256 + threadIdx.x;
    if (i < n) dst[i] = (__bf16)src[i];
}

// ---------------------------------------------------------------------------
// 2) Activations: act[:, 0:1024] = silu(x); act[:, 1024:] = degree-3 b-splines
// ---------------------------------------------------------------------------
__global__ void act_kernel(const float* __restrict__ x,
                           const float* __restrict__ grid,
                           __bf16* __restrict__ act) {
    int i = blockIdx.x * 256 + threadIdx.x;
    if (i >= SEQ * HID) return;
    int s = i >> 10, f = i & 1023;
    float xv = x[i];
    __bf16* row = act + (size_t)s * KCAT;
    row[f] = (__bf16)(xv / (1.f + __expf(-xv)));

    float g[12];
#pragma unroll
    for (int j = 0; j < 12; ++j) g[j] = grid[f * 12 + j];
    float b[11];
#pragma unroll
    for (int j = 0; j < 11; ++j)
        b[j] = (xv >= g[j] && xv < g[j + 1]) ? 1.f : 0.f;
#pragma unroll
    for (int k = 1; k <= 3; ++k) {
        for (int j = 0; j <= 10 - k; ++j) {
            float left  = (xv - g[j]) / (g[j + k] - g[j]);
            float right = (g[j + k + 1] - xv) / (g[j + k + 1] - g[j + 1]);
            b[j] = left * b[j] + right * b[j + 1];
        }
    }
    __bf16* sp = row + HID + f * NCOEF;
#pragma unroll
    for (int c = 0; c < NCOEF; ++c) sp[c] = (__bf16)b[c];
}

// ---------------------------------------------------------------------------
// 3/6) WMMA bf16 GEMM with async-LDS-staged weights.
//   Block = 256 thr (8 waves) -> C tile 256(M) x 64(N); wave tile 32x64.
//   Per k-step: block stages W[64x32] (4KB) via one async b128/thread,
//   double-buffered; each wave does 8 WMMAs (2 A-frags x 4 shared B-frags).
// ---------------------------------------------------------------------------
__global__ void __launch_bounds__(256)
gemm_bf16_staged(const __bf16* __restrict__ A,
                 const __bf16* __restrict__ W,
                 const float* __restrict__ bias,
                 float* __restrict__ C,
                 int M, int N, int K) {
    __shared__ __bf16 Wt[2][64 * 32];
    int tid  = threadIdx.x;
    int w    = tid >> 5;
    int lane = tid & 31;
    int nt   = N >> 6;
    int mblk = blockIdx.x / nt, nblk = blockIdx.x % nt;
    int n0   = nblk << 6;
    int m0   = mblk * 256 + w * 32;

    // staging assignment: thread -> (row tn of 64, 8-element k segment)
    int tn = tid >> 2;
    int tk = (tid & 3) << 3;
    const __bf16* gW = W + (size_t)(n0 + tn) * K + tk;

    stage16(&Wt[0][tn * 32 + tk], gW);           // prologue: k0 = 0 -> buf 0

    AccF acc[8] = {};
    for (int k0 = 0; k0 < K; k0 += 32) {
        int buf = (k0 >> 5) & 1;
        async_wait0();
        __syncthreads();                         // staged tile visible block-wide
        if (k0 + 32 < K)
            stage16(&Wt[buf ^ 1][tn * 32 + tk], gW + (k0 + 32));

        v16bf a0 = load_a_frag(A, K, m0,      k0);
        v16bf a1 = load_a_frag(A, K, m0 + 16, k0);
#pragma unroll
        for (int c = 0; c < 4; ++c) {
            v16bf b = load_b_fragT(&Wt[buf][0], 32, c * 16, 0);
            acc[c].v     = wmma_bf16(a0, b, acc[c].v);
            acc[4 + c].v = wmma_bf16(a1, b, acc[4 + c].v);
        }
    }

    int n = lane & 15, h = lane >> 4;
#pragma unroll
    for (int half = 0; half < 2; ++half)
#pragma unroll
        for (int c = 0; c < 4; ++c)
#pragma unroll
            for (int vi = 0; vi < 8; ++vi) {
                int m   = m0 + half * 16 + h * 8 + vi;
                int col = n0 + c * 16 + n;
                float val = acc[half * 4 + c].f[vi];
                if (bias) val += bias[col];
                C[(size_t)m * N + col] = val;
            }
}

// ---------------------------------------------------------------------------
// 4) RoPE + head-major repack (scale 1/8 folded into Q)
// ---------------------------------------------------------------------------
__global__ void rope_kernel(const float* __restrict__ qkv,
                            const float* __restrict__ rc,
                            const float* __restrict__ rs,
                            __bf16* __restrict__ Q,
                            __bf16* __restrict__ Kd,
                            __bf16* __restrict__ V) {
    int i = blockIdx.x * 256 + threadIdx.x;
    if (i >= SEQ * NHEAD * 32) return;
    int t  = i >> 9;
    int r  = i & 511;
    int hd = r >> 5;
    int p  = r & 31;
    const float* base = qkv + (size_t)t * OUTF + hd * (3 * HDIM);
    float c  = rc[t * 32 + p];
    float sn = rs[t * 32 + p];
    size_t off = ((size_t)hd * SEQ + t) * HDIM + 2 * p;

    float qr = base[2 * p], qi = base[2 * p + 1];
    Q[off]     = (__bf16)(0.125f * (qr * c - qi * sn));
    Q[off + 1] = (__bf16)(0.125f * (qr * sn + qi * c));

    float kr = base[HDIM + 2 * p], ki = base[HDIM + 2 * p + 1];
    Kd[off]     = (__bf16)(kr * c - ki * sn);
    Kd[off + 1] = (__bf16)(kr * sn + ki * c);

    V[off]     = (__bf16)base[2 * HDIM + 2 * p];
    V[off + 1] = (__bf16)base[2 * HDIM + 2 * p + 1];
}

// ---------------------------------------------------------------------------
// 5) Flash attention: one wave per (head, 16-query tile), 4 waves/block.
// ---------------------------------------------------------------------------
__global__ void __launch_bounds__(128)
attn_kernel(const __bf16* __restrict__ Q,
            const __bf16* __restrict__ Kd,
            const __bf16* __restrict__ V,
            __bf16* __restrict__ ctx) {
    __shared__ __bf16 Pbuf[4][16 * 32];
    int wl   = threadIdx.x >> 5;
    int wave = blockIdx.x * 4 + wl;
    int lane = threadIdx.x & 31;
    int head = wave >> 7;
    int qt   = wave & 127;
    int q0   = qt * 16;
    int n = lane & 15, h = lane >> 4;

    const __bf16* Qh = Q  + (size_t)head * SEQ * HDIM;
    const __bf16* Kh = Kd + (size_t)head * SEQ * HDIM;
    const __bf16* Vh = V  + (size_t)head * SEQ * HDIM;

    v16bf qa0 = load_a_frag(Qh, HDIM, q0, 0);
    v16bf qa1 = load_a_frag(Qh, HDIM, q0, 32);

    AccF o[4] = {};
    float mrun[8], lrun[8];
#pragma unroll
    for (int vi = 0; vi < 8; ++vi) { mrun[vi] = -1e30f; lrun[vi] = 0.f; }

    __bf16* Pl = &Pbuf[wl][0];

    for (int kt = 0; kt < SEQ; kt += 32) {
        AccF s0 = {}, s1 = {};
        v16bf b;
        b = load_b_fragT(Kh, HDIM, kt,      0);  s0.v = wmma_bf16(qa0, b, s0.v);
        b = load_b_fragT(Kh, HDIM, kt,      32); s0.v = wmma_bf16(qa1, b, s0.v);
        b = load_b_fragT(Kh, HDIM, kt + 16, 0);  s1.v = wmma_bf16(qa0, b, s1.v);
        b = load_b_fragT(Kh, HDIM, kt + 16, 32); s1.v = wmma_bf16(qa1, b, s1.v);

#pragma unroll
        for (int vi = 0; vi < 8; ++vi) {
            float rm = fmaxf(s0.f[vi], s1.f[vi]);
            rm = fmaxf(rm, __shfl_xor(rm, 1));
            rm = fmaxf(rm, __shfl_xor(rm, 2));
            rm = fmaxf(rm, __shfl_xor(rm, 4));
            rm = fmaxf(rm, __shfl_xor(rm, 8));
            float mn    = fmaxf(mrun[vi], rm);
            float alpha = __expf(mrun[vi] - mn);
            mrun[vi] = mn;
            float p0 = __expf(s0.f[vi] - mn);
            float p1 = __expf(s1.f[vi] - mn);
            float rsum = p0 + p1;
            rsum += __shfl_xor(rsum, 1);
            rsum += __shfl_xor(rsum, 2);
            rsum += __shfl_xor(rsum, 4);
            rsum += __shfl_xor(rsum, 8);
            lrun[vi] = lrun[vi] * alpha + rsum;
#pragma unroll
            for (int c = 0; c < 4; ++c) o[c].f[vi] *= alpha;
            int m = h * 8 + vi;
            Pl[m * 32 + n]      = (__bf16)p0;
            Pl[m * 32 + 16 + n] = (__bf16)p1;
        }
        v16bf pa = load_a_frag(Pl, 32, 0, 0);   // same-wave LDS RAW: DS in-order
#pragma unroll
        for (int c = 0; c < 4; ++c) {
            v16bf vb = load_b_fragN(Vh, HDIM, kt, c * 16);
            o[c].v = wmma_bf16(pa, vb, o[c].v);
        }
    }

#pragma unroll
    for (int vi = 0; vi < 8; ++vi) {
        float inv = 1.f / lrun[vi];
        int m = q0 + h * 8 + vi;
#pragma unroll
        for (int c = 0; c < 4; ++c)
            ctx[(size_t)m * HID + head * HDIM + c * 16 + n] = (__bf16)(o[c].f[vi] * inv);
    }
}

// ---------------------------------------------------------------------------
// Launch
// ---------------------------------------------------------------------------
extern "C" void kernel_launch(void* const* d_in, const int* in_sizes, int n_in,
                              void* d_out, int out_size, void* d_ws, size_t ws_size,
                              hipStream_t stream) {
    const float* x       = (const float*)d_in[0];
    const float* bw      = (const float*)d_in[1];
    const float* sw      = (const float*)d_in[2];
    const float* scaler  = (const float*)d_in[3];
    const float* out_w   = (const float*)d_in[4];
    const float* out_b   = (const float*)d_in[5];
    const float* grid    = (const float*)d_in[6];
    const float* rot_cos = (const float*)d_in[7];
    const float* rot_sin = (const float*)d_in[8];
    float* out = (float*)d_out;

    char* ws = (char*)d_ws;
    size_t off = 0;
    __bf16* wcat  = (__bf16*)(ws + off); off += (size_t)OUTF * KCAT * 2;
    __bf16* outwb = (__bf16*)(ws + off); off += (size_t)HID * HID * 2;
    __bf16* act   = (__bf16*)(ws + off); off += (size_t)SEQ * KCAT * 2;
    float*  qkv   = (float*) (ws + off); off += (size_t)SEQ * OUTF * 4;
    __bf16* Qb    = (__bf16*)(ws + off); off += (size_t)NHEAD * SEQ * HDIM * 2;
    __bf16* Kb    = (__bf16*)(ws + off); off += (size_t)NHEAD * SEQ * HDIM * 2;
    __bf16* Vb    = (__bf16*)(ws + off); off += (size_t)NHEAD * SEQ * HDIM * 2;
    __bf16* ctxb  = (__bf16*)(ws + off); off += (size_t)SEQ * HID * 2;

    {
        size_t tot = (size_t)OUTF * KCAT;
        pack_wcat_kernel<<<(unsigned)((tot + 255) / 256), 256, 0, stream>>>(bw, sw, scaler, wcat);
        pack_bf16_kernel<<<(HID * HID + 255) / 256, 256, 0, stream>>>(out_w, outwb, HID * HID);
    }
    act_kernel<<<(SEQ * HID + 255) / 256, 256, 0, stream>>>(x, grid, act);

    // fused base+spline GEMM: M=2048, N=3072, K=9216 ; 8 x 48 = 384 blocks
    gemm_bf16_staged<<<(SEQ / 256) * (OUTF / 64), 256, 0, stream>>>(
        act, wcat, nullptr, qkv, SEQ, OUTF, KCAT);

    rope_kernel<<<(SEQ * NHEAD * 32 + 255) / 256, 256, 0, stream>>>(qkv, rot_cos, rot_sin, Qb, Kb, Vb);

    attn_kernel<<<(NHEAD * 128) / 4, 128, 0, stream>>>(Qb, Kb, Vb, ctxb);

    // output projection: M=2048, N=1024, K=1024 ; 8 x 16 = 128 blocks
    gemm_bf16_staged<<<(SEQ / 256) * (HID / 64), 256, 0, stream>>>(
        ctxb, outwb, out_b, out, SEQ, HID, HID);
}
